// MultiHeadedAttention_64544768524337
// MI455X (gfx1250) — compile-verified, matrix operand order checked by reading, not run on hardware
//
#include <hip/hip_runtime.h>
#include <hip/hip_bf16.h>

typedef _Float16 f16;
typedef __attribute__((ext_vector_type(16))) _Float16 v16h;
typedef __attribute__((ext_vector_type(8)))  float    v8f;
typedef __attribute__((ext_vector_type(4)))  int      v4i;

#define DM 1024   // d_model
#define NN 1024   // sequence length
#define BATCH 8
#define NH 16     // heads
#define HD 64     // head dim

#define BM 128
#define BN 128
#define BK 32

union Frag16 { v16h v; unsigned u[8]; f16 h[16]; };
union Ld32   { float4 q[2]; f16 h[16]; };
union Ld16   { float4 q;    f16 h[8];  };

// ---------------------------------------------------------------------------
// Async global->LDS copy (16 bytes per lane), tracked on ASYNCcnt.
// Builtin signature (from compiler diagnostic): param0 is v4i in AS(1).
// Falls back to a synchronous 16B copy if the builtin is not declared.
// ---------------------------------------------------------------------------
#if defined(__gfx1250__) && __has_builtin(__builtin_amdgcn_global_load_async_to_lds_b128)
#define USE_ASYNC_LDS 1
#else
#define USE_ASYNC_LDS 0
#endif

__device__ inline void async_copy16(const f16* g, f16* l) {
#if USE_ASYNC_LDS
  f16* gnc = (f16*)g;  // strip const first; then retype + addrspacecast
  __builtin_amdgcn_global_load_async_to_lds_b128(
      (v4i __attribute__((address_space(1)))*)gnc,
      (v4i __attribute__((address_space(3)))*)l, 0, 0);
#else
  *(float4*)l = *(const float4*)g;
#endif
}

__device__ inline void wait_async() {
#if USE_ASYNC_LDS
#if __has_builtin(__builtin_amdgcn_s_wait_asynccnt)
  __builtin_amdgcn_s_wait_asynccnt(0);
#else
  asm volatile("s_wait_asynccnt 0" ::: "memory");
#endif
#endif
}

// ---------------------------------------------------------------------------
// fp32 -> f16 conversion (grid-stride)
// ---------------------------------------------------------------------------
__global__ void cvt_f32_to_f16(const float* __restrict__ in, f16* __restrict__ out, long n) {
  long i = (long)blockIdx.x * blockDim.x + threadIdx.x;
  long stride = (long)gridDim.x * blockDim.x;
  for (; i < n; i += stride) out[i] = (f16)in[i];
}

// ---------------------------------------------------------------------------
// Build a 16x32 f16 A/B fragment from an LDS tile stored K-contiguous.
// CDNA5 16-bit layout: lanes 16-31 add +8 to K, vector halves 8..15 add +16,
// pairs are K-consecutive -> dword loads (coalesce to ds_load_b128).
// ---------------------------------------------------------------------------
__device__ inline v16h frag_from_lds(const f16* rowbase, int khalf) {
  Frag16 f;
  const unsigned* p32 = (const unsigned*)rowbase;
#pragma unroll
  for (int p = 0; p < 8; ++p) {
    int k = khalf + ((p & 4) << 2) + ((p & 3) << 1);
    f.u[p] = p32[k >> 1];
  }
  return f.v;
}

// ---------------------------------------------------------------------------
// GEMM: Y[b] = W(1024x1024) @ X[b](1024x1024) + bias   (per-batch, f16 WMMA)
// Double-buffered LDS; W tile filled with async-to-LDS, X tile transposed.
// writeF32 ? write float to Yf : write f16 to Yh
// ---------------------------------------------------------------------------
__global__ __launch_bounds__(256)
void gemm1024_wmma(const f16* __restrict__ W, const f16* __restrict__ X,
                   const float* __restrict__ bias,
                   f16* __restrict__ Yh, float* __restrict__ Yf, int writeF32)
{
  __shared__ f16 ldsW[2][BM * BK];   // [m][k], k contiguous
  __shared__ f16 ldsX[2][BN * BK];   // [n][k] (transposed), k contiguous

  const int b  = blockIdx.z;
  const f16* Xb = X + (size_t)b * DM * NN;
  const int m0 = blockIdx.y * BM;
  const int n0 = blockIdx.x * BN;
  const int tid  = threadIdx.x;
  const int lane = tid & 31;
  const int wave = tid >> 5;
  const int wm = (wave >> 1) * 32;   // 4 waves along M
  const int wn = (wave & 1) * 64;    // 2 waves along N
  const int lm = lane & 15;
  const int hi = lane >> 4;
  const int khalf = hi * 8;

  auto stageW = [&](int buf, int k0) {
    int m  = tid >> 1;
    int kk = (tid & 1) * 16;
    const f16* src = W + (size_t)(m0 + m) * DM + k0 + kk;
    f16* dst = &ldsW[buf][m * BK + kk];
    async_copy16(src, dst);
    async_copy16(src + 8, dst + 8);
  };
  auto stageX = [&](int buf, int k0) {
    int kk = tid >> 3;          // 0..31
    int nn = (tid & 7) * 16;    // 0..112
    Ld32 t;
    const float4* s4 = (const float4*)(Xb + (size_t)(k0 + kk) * NN + n0 + nn);
    t.q[0] = s4[0]; t.q[1] = s4[1];
#pragma unroll
    for (int j = 0; j < 16; ++j) ldsX[buf][(nn + j) * BK + kk] = t.h[j];
    if (k0 + BK < DM)
      __builtin_prefetch(Xb + (size_t)(k0 + BK + kk) * NN + n0 + nn, 0, 1);
  };

  v8f zero = {};
  v8f acc[2][4];
#pragma unroll
  for (int i = 0; i < 2; ++i)
#pragma unroll
    for (int j = 0; j < 4; ++j) acc[i][j] = zero;

  // prologue: fill buffer 0
  stageW(0, 0);
  stageX(0, 0);
  wait_async();
  __syncthreads();

  for (int k0 = 0; k0 < DM; k0 += BK) {
    const int cur = (k0 >> 5) & 1;
    const int nxt = cur ^ 1;

    // overlap: start filling the next tile before computing on the current
    if (k0 + BK < DM) {
      stageW(nxt, k0 + BK);
      stageX(nxt, k0 + BK);
    }

    v16h a[2], bb[4];
#pragma unroll
    for (int mt = 0; mt < 2; ++mt)
      a[mt]  = frag_from_lds(&ldsW[cur][(wm + mt * 16 + lm) * BK], khalf);
#pragma unroll
    for (int nt = 0; nt < 4; ++nt)
      bb[nt] = frag_from_lds(&ldsX[cur][(wn + nt * 16 + lm) * BK], khalf);

#pragma unroll
    for (int mt = 0; mt < 2; ++mt)
#pragma unroll
      for (int nt = 0; nt < 4; ++nt)
        acc[mt][nt] = __builtin_amdgcn_wmma_f32_16x16x32_f16(
            false, a[mt], false, bb[nt], (short)0, acc[mt][nt], false, false);

    wait_async();
    __syncthreads();
  }

  // epilogue: C/D layout row = vgpr + 8*hi, col = lane&15
#pragma unroll
  for (int mt = 0; mt < 2; ++mt) {
#pragma unroll
    for (int nt = 0; nt < 4; ++nt) {
      int col = n0 + wn + nt * 16 + lm;
#pragma unroll
      for (int r = 0; r < 8; ++r) {
        int row = m0 + wm + mt * 16 + hi * 8 + r;
        float v = acc[mt][nt][r] + bias[row];
        size_t idx = ((size_t)b * DM + row) * NN + col;
        if (writeF32) Yf[idx] = v;
        else          Yh[idx] = (f16)v;
      }
    }
  }
}

// ---------------------------------------------------------------------------
// Flash attention over Q/K/V f16 buffers laid out [b, c, n], c = d*16 + h.
// One block handles one (b, h) and one 128-query chunk; its 8 waves cover 8 query tiles of 16 rows
// and SHARE the K/V tiles staged cooperatively into LDS per 32-key step.
// ---------------------------------------------------------------------------
__global__ __launch_bounds__(256)
void attn_wmma(const f16* __restrict__ Q, const f16* __restrict__ Km,
               const f16* __restrict__ Vm, f16* __restrict__ Xa)
{
  __shared__ f16 ldsKt[32 * 64];     // K tile transposed: [m][d], d contiguous
  __shared__ f16 ldsV [64 * 32];     // V tile:            [d][m], m contiguous
  __shared__ f16 ldsP [8][16 * 32];  // per-wave P transpose slab

  const int tid  = threadIdx.x;
  const int lane = tid & 31;
  const int wave = tid >> 5;
  const int lm = lane & 15;
  const int hi = lane >> 4;

  const int gid = blockIdx.x;        // B*H*8 = 1024 blocks
  const int b  = gid >> 7;
  const int h  = (gid >> 3) & 15;
  const int qc = gid & 7;
  const int n0 = (qc * 8 + wave) * 16;

  const size_t boff = (size_t)b * DM * NN;
  const f16* Qb = Q  + boff;
  const f16* Kb = Km + boff;
  const f16* Vb = Vm + boff;

  // Q^T as A fragments: rows n (lane), K = head-dim d (64 -> 2 frags)
  Frag16 qa[2];
#pragma unroll
  for (int f = 0; f < 2; ++f)
#pragma unroll
    for (int e = 0; e < 16; ++e) {
      int d = f * 32 + hi * 8 + ((e & 8) << 1) + (e & 7);
      qa[f].h[e] = Qb[(size_t)(d * NH + h) * NN + (n0 + lm)];
    }

  v8f zero = {};
  v8f oacc[4];
#pragma unroll
  for (int f = 0; f < 4; ++f) oacc[f] = zero;
  float mrow[8], lrow[8];
#pragma unroll
  for (int r = 0; r < 8; ++r) { mrow[r] = -1.0e30f; lrow[r] = 0.0f; }

  f16* pl = &ldsP[wave][0];
  const int sd = tid >> 2;           // 0..63: head-dim row for staging
  const int sm = (tid & 3) * 8;      // 0,8,16,24: key chunk for staging

  for (int mS = 0; mS < NN; mS += 32) {
    __syncthreads();                 // previous iteration's LDS reads done

    // ---- stage V tile [d][m] (straight copy -> async-to-LDS)
    async_copy16(Vb + (size_t)(sd * NH + h) * NN + mS + sm, &ldsV[sd * 32 + sm]);

    // ---- stage K tile transposed to [m][d]
    {
      Ld16 t4;
      t4.q = *(const float4*)(Kb + (size_t)(sd * NH + h) * NN + mS + sm);
#pragma unroll
      for (int j = 0; j < 8; ++j) ldsKt[(sm + j) * 64 + sd] = t4.h[j];
    }
    wait_async();
    __syncthreads();

    // ---- S = scale * Q^T K  (two 16x16 tiles covering 32 keys)
    v8f s[2];
#pragma unroll
    for (int t = 0; t < 2; ++t) {
      s[t] = zero;
#pragma unroll
      for (int f = 0; f < 2; ++f) {
        Frag16 kb;
        const unsigned* p32 = (const unsigned*)(ldsKt + (t * 16 + lm) * 64);
#pragma unroll
        for (int p = 0; p < 8; ++p) {
          int k = f * 32 + hi * 8 + ((p & 4) << 2) + ((p & 3) << 1);
          kb.u[p] = p32[k >> 1];
        }
        s[t] = __builtin_amdgcn_wmma_f32_16x16x32_f16(
            false, qa[f].v, false, kb.v, (short)0, s[t], false, false);
      }
    }

    // ---- online softmax (rows r+8*hi; cols live across 16-lane halves)
    float p0[8], p1[8];
#pragma unroll
    for (int r = 0; r < 8; ++r) {
      float s0 = s[0][r] * 0.125f;   // 1/sqrt(64)
      float s1 = s[1][r] * 0.125f;
      float rmax = fmaxf(s0, s1);
#pragma unroll
      for (int msk = 1; msk < 16; msk <<= 1)
        rmax = fmaxf(rmax, __shfl_xor(rmax, msk, 32));
      float mnew  = fmaxf(mrow[r], rmax);
      float alpha = __expf(mrow[r] - mnew);
      mrow[r] = mnew;
      p0[r] = __expf(s0 - mnew);
      p1[r] = __expf(s1 - mnew);
      float rs = p0[r] + p1[r];
#pragma unroll
      for (int msk = 1; msk < 16; msk <<= 1)
        rs += __shfl_xor(rs, msk, 32);
      lrow[r] = lrow[r] * alpha + rs;
#pragma unroll
      for (int f = 0; f < 4; ++f) oacc[f][r] *= alpha;
    }

    // ---- transpose P (C layout) -> A layout via per-wave LDS slab
#pragma unroll
    for (int r = 0; r < 8; ++r) {
      pl[(r + 8 * hi) * 32 + lm]      = (f16)p0[r];
      pl[(r + 8 * hi) * 32 + lm + 16] = (f16)p1[r];
    }
    __syncthreads();
    Frag16 pa;
    {
      const unsigned* p32 = (const unsigned*)(pl + lm * 32);
#pragma unroll
      for (int p = 0; p < 8; ++p) {
        int k = hi * 8 + ((p & 4) << 2) + ((p & 3) << 1);
        pa.u[p] = p32[k >> 1];
      }
    }

    // ---- O^T += P V^T   (K = 32 keys, 4 tiles covering d=64)
#pragma unroll
    for (int f = 0; f < 4; ++f) {
      Frag16 vb;
      const unsigned* p32 = (const unsigned*)(ldsV + (f * 16 + lm) * 32);
#pragma unroll
      for (int p = 0; p < 8; ++p) {
        int m = hi * 8 + ((p & 4) << 2) + ((p & 3) << 1);
        vb.u[p] = p32[m >> 1];
      }
      oacc[f] = __builtin_amdgcn_wmma_f32_16x16x32_f16(
          false, pa.v, false, vb.v, (short)0, oacc[f], false, false);
    }
  }

  // ---- normalize and scatter O^T[n,d] back to [b, d*16+h, n] (f16)
#pragma unroll
  for (int f = 0; f < 4; ++f) {
    int d = f * 16 + lm;
#pragma unroll
    for (int r = 0; r < 8; ++r) {
      int n = n0 + 8 * hi + r;
      float v = oacc[f][r] / lrow[r];
      Xa[boff + (size_t)(d * NH + h) * NN + n] = (f16)v;
    }
  }
}

// ---------------------------------------------------------------------------
extern "C" void kernel_launch(void* const* d_in, const int* in_sizes, int n_in,
                              void* d_out, int out_size, void* d_ws, size_t ws_size,
                              hipStream_t stream) {
  const float* q_in = (const float*)d_in[0];
  const float* k_in = (const float*)d_in[1];
  const float* v_in = (const float*)d_in[2];
  const float* Wq = (const float*)d_in[3];
  const float* bq = (const float*)d_in[4];
  const float* Wk = (const float*)d_in[5];
  const float* bk = (const float*)d_in[6];
  const float* Wv = (const float*)d_in[7];
  const float* bv = (const float*)d_in[8];
  const float* Wm = (const float*)d_in[9];
  const float* bm = (const float*)d_in[10];
  float* out = (float*)d_out;

  // workspace partition (all f16)
  f16* ws = (f16*)d_ws;
  const size_t SB = (size_t)BATCH * DM * NN;  // 8M elems
  const size_t SW = (size_t)DM * DM;          // 1M elems
  f16* q16  = ws;
  f16* k16  = q16  + SB;
  f16* v16  = k16  + SB;
  f16* wq16 = v16  + SB;
  f16* wk16 = wq16 + SW;
  f16* wv16 = wk16 + SW;
  f16* wm16 = wv16 + SW;
  f16* Qp   = wm16 + SW;
  f16* Kp   = Qp   + SB;
  f16* Vp   = Kp   + SB;
  f16* Xat  = Vp   + SB;

  // 1) convert fp32 inputs/weights to f16
  cvt_f32_to_f16<<<1024, 256, 0, stream>>>(q_in, q16, (long)SB);
  cvt_f32_to_f16<<<1024, 256, 0, stream>>>(k_in, k16, (long)SB);
  cvt_f32_to_f16<<<1024, 256, 0, stream>>>(v_in, v16, (long)SB);
  cvt_f32_to_f16<<<256, 256, 0, stream>>>(Wq, wq16, (long)SW);
  cvt_f32_to_f16<<<256, 256, 0, stream>>>(Wk, wk16, (long)SW);
  cvt_f32_to_f16<<<256, 256, 0, stream>>>(Wv, wv16, (long)SW);
  cvt_f32_to_f16<<<256, 256, 0, stream>>>(Wm, wm16, (long)SW);

  // 2) Q/K/V projections (WMMA GEMM, f16 out)
  dim3 gg(NN / BN, DM / BM, BATCH);
  gemm1024_wmma<<<gg, 256, 0, stream>>>(wq16, q16, bq, Qp, nullptr, 0);
  gemm1024_wmma<<<gg, 256, 0, stream>>>(wk16, k16, bk, Kp, nullptr, 0);
  gemm1024_wmma<<<gg, 256, 0, stream>>>(wv16, v16, bv, Vp, nullptr, 0);

  // 3) flash attention: 8 blocks per (b,h), 128 queries each; 8 waves share K/V LDS tiles
  attn_wmma<<<BATCH * NH * (NN / 128), 256, 0, stream>>>(Qp, Kp, Vp, Xat);

  // 4) output projection (fp32 out)
  gemm1024_wmma<<<gg, 256, 0, stream>>>(wm16, Xat, bm, nullptr, out, 1);
}
